// ImportanceRenderer_60576218743096
// MI455X (gfx1250) — matrix operand order repaired and checked by reading, not compiled, not run on hardware
//
#include <hip/hip_runtime.h>
#include <math.h>

// ---- problem constants (from reference) ----
#define PLN   3        // P: planes
#define LL    6        // L: levels
#define CPL   32       // channels per level
#define HID   64       // hidden
#define HP    256      // plane H=W
#define NPTS  65536    // M points
#define NB    2        // N batches
#define OUTC  17       // sdf,sigma,rgb(3),sem(6),grad(3),grad_cano(3)

#define PTS   64       // points per block
#define BLOCK 128      // 4 waves of 32

typedef _Float16 v16h __attribute__((ext_vector_type(16)));
typedef _Float16 v8h  __attribute__((ext_vector_type(8)));
typedef float    v8f  __attribute__((ext_vector_type(8)));

struct Bil { int x0, y0; float wx, wy; };

__device__ __forceinline__ float tapv(const float* __restrict__ img, int x, int y) {
    // zeros padding (matches grid_sample align_corners=False, padding=zeros)
    return (x >= 0 && x < HP && y >= 0 && y < HP) ? img[y * HP + x] : 0.f;
}

__device__ __forceinline__ v16h cat16(v8h lo, v8h hi) {
    return __builtin_shufflevector(lo, hi, 0, 1, 2, 3, 4, 5, 6, 7,
                                           8, 9, 10, 11, 12, 13, 14, 15);
}

// softplus via hw transcendentals: max(z,0) + log(1+exp(-|z|))
__device__ __forceinline__ float softplus_fast(float z) {
    return fmaxf(z, 0.f) + __logf(1.f + __expf(-fabsf(z)));
}

__global__ __launch_bounds__(BLOCK) void triplane_fused_kernel(
    const float* __restrict__ planes,   // [NB,PLN,LL*CPL,HP,HP]
    const float* __restrict__ coords,   // [NB,NPTS,3]
    const float* __restrict__ w1,       // [LL,CPL,HID]
    const float* __restrict__ b1,       // [LL,HID]
    const float* __restrict__ w2,       // [LL,HID,5]
    const float* __restrict__ b2,       // [LL,5]
    const float* __restrict__ betap,    // [1]
    float* __restrict__ out)            // [NB,NPTS,OUTC]
{
    __shared__ __align__(16) _Float16 featS[PTS][LL * CPL]; // 24 KB gathered features
    __shared__ __align__(16) _Float16 hS[PTS][HID];         //  8 KB layer-1 activations
    __shared__ __align__(16) _Float16 w1S[CPL][HID];        //  4 KB per-l (rows: k-cpl)
    __shared__ __align__(16) _Float16 w1T[HID][CPL];        //  4 KB per-l transposed
    __shared__ __align__(16) _Float16 w2T[16][HID];         //  2 KB per-l transposed+pad
    __shared__ float    gfS[PTS][CPL];        //  8 KB  d(delta)/d(feat), current l
    __shared__ float    sigS[LL][PTS];
    __shared__ float    rgbS[LL][PTS][3];
    __shared__ float    gcS[PTS][3];
    __shared__ Bil      blS[PTS][PLN];
    __shared__ float    w2c0S[HID];
    __shared__ float    b1S[HID];
    __shared__ float    b2S[8];

    const int tid  = threadIdx.x;
    const int lane = tid & 31;
    const int wv   = tid >> 5;                      // wave 0..3, owns 16 points
    const int n    = blockIdx.x / (NPTS / PTS);
    const int m0   = (blockIdx.x % (NPTS / PTS)) * PTS;
    const float beta = betap[0];

    // WMMA fragment index helpers (ISA 7.12.2, wave32):
    const int row     = lane & 15;                  // A: M-row / B,C: N-col
    const int kbaseA  = (lane < 16) ? 0 : 8;        // A 16x32 f16: halves -> K
    const int kbaseB  = (lane < 16) ? 0 : 8;        // B 32x16 f16: halves -> K (same order as A)
    const int mrowOff = (lane < 16) ? 0 : 8;        // C 16x16 f32: vgpr v -> M

    // ---- phase 0: per-point bilinear params + grad accumulator init ----
    for (int i = tid; i < PTS * PLN; i += BLOCK) {
        int pt = i / PLN, p = i % PLN;
        const float* c = coords + ((size_t)n * NPTS + m0 + pt) * 3;
        float cx = c[0], cy = c[1], cz = c[2];
        // scaled = coords * (2/BOX_WARP) = coords; proj = axis permutation
        float gx, gy;
        if (p == 0)      { gx = cx; gy = cy; }
        else if (p == 1) { gx = cx; gy = cz; }
        else             { gx = cz; gy = cy; }
        float x = (gx + 1.f) * (HP * 0.5f) - 0.5f;
        float y = (gy + 1.f) * (HP * 0.5f) - 0.5f;
        float x0f = floorf(x), y0f = floorf(y);
        Bil b; b.x0 = (int)x0f; b.y0 = (int)y0f; b.wx = x - x0f; b.wy = y - y0f;
        blS[pt][p] = b;
    }
    for (int i = tid; i < PTS * 3; i += BLOCK) gcS[i / 3][i % 3] = 0.f;
    __syncthreads();

    // ---- phase 1: gather features (bilinear, mean over planes) -> LDS f16 ----
    {
        int pt  = tid >> 1;                     // 2 threads per point
        int ch0 = (tid & 1) * (LL * CPL / 2);   // 96 channels each
        Bil bls[PLN];
        #pragma unroll
        for (int p = 0; p < PLN; ++p) bls[p] = blS[pt][p];
        for (int c = 0; c < LL * CPL / 2; ++c) {
            int ch = ch0 + c;
            float acc = 0.f;
            #pragma unroll
            for (int p = 0; p < PLN; ++p) {
                Bil b = bls[p];
                const float* img = planes + (((size_t)(n * PLN + p) * (LL * CPL) + ch) * (HP * HP));
                float v00 = tapv(img, b.x0,     b.y0);
                float v10 = tapv(img, b.x0 + 1, b.y0);
                float v01 = tapv(img, b.x0,     b.y0 + 1);
                float v11 = tapv(img, b.x0 + 1, b.y0 + 1);
                acc += (v00 * (1.f - b.wx) + v10 * b.wx) * (1.f - b.wy)
                     + (v01 * (1.f - b.wx) + v11 * b.wx) * b.wy;
            }
            featS[pt][ch] = (_Float16)(acc * (1.f / 3.f));
        }
    }

    // ---- per-level MLP forward + backward (WMMA) + gradient re-gather ----
    for (int l = 0; l < LL; ++l) {
        __syncthreads();  // protect per-l weight LDS & gfS reuse
        for (int i = tid; i < CPL * HID; i += BLOCK) {
            int c = i / HID, hh = i % HID;
            _Float16 v = (_Float16)w1[(size_t)l * CPL * HID + i];
            w1S[c][hh] = v;
            w1T[hh][c] = v;
        }
        for (int i = tid; i < HID * 16; i += BLOCK) {
            int k = i >> 4, j = i & 15;
            w2T[j][k] = (_Float16)((j < 5) ? w2[((size_t)l * HID + k) * 5 + j] : 0.f);
        }
        for (int i = tid; i < HID; i += BLOCK) {
            w2c0S[i] = w2[((size_t)l * HID + i) * 5];
            b1S[i]   = b1[l * HID + i];
        }
        if (tid < 5) b2S[tid] = b2[l * 5 + tid];
        __syncthreads();

        // layer 1: feat[16x32] x w1[32x64]  (K=32, one WMMA step per N-chunk)
        const _Float16* fr = &featS[wv * 16 + row][l * CPL];
        v16h a = cat16(*(const v8h*)(fr + kbaseA),
                       *(const v8h*)(fr + kbaseA + 16));
        #pragma unroll
        for (int nc = 0; nc < 4; ++nc) {
            const _Float16* br = &w1T[nc * 16 + row][kbaseB];  // K runs kbaseB..+7, kbaseB+16..+23
            v16h bfrag = cat16(*(const v8h*)(br), *(const v8h*)(br + 16));
            v8f c = {};
            c = __builtin_amdgcn_wmma_f32_16x16x32_f16(false, a, false, bfrag,
                                                       (short)0, c, false, false);
            #pragma unroll
            for (int v = 0; v < 8; ++v) {          // bias + softplus -> h (f16)
                int mm  = v + mrowOff;
                int col = nc * 16 + row;
                float z = c[v] + b1S[col];
                hS[wv * 16 + mm][col] = (_Float16)softplus_fast(z);
            }
        }

        // layer 2: h[16x64] x w2p[64x16]  (2 K-steps, cols 0..4 real)
        v8f oc = {};
        #pragma unroll
        for (int s = 0; s < 2; ++s) {
            const _Float16* hr = &hS[wv * 16 + row][s * 32 + kbaseA];
            v16h a2 = cat16(*(const v8h*)(hr), *(const v8h*)(hr + 16));
            const _Float16* wr = &w2T[row][s * 32 + kbaseB];
            v16h bf = cat16(*(const v8h*)(wr), *(const v8h*)(wr + 16));
            oc = __builtin_amdgcn_wmma_f32_16x16x32_f16(false, a2, false, bf,
                                                        (short)0, oc, false, false);
        }
        #pragma unroll
        for (int v = 0; v < 8; ++v) {
            int pt = wv * 16 + v + mrowOff;
            if (row == 0)      sigS[l][pt]          = oc[v] + b2S[0];
            else if (row <= 3) rgbS[l][pt][row - 1] = oc[v] + b2S[row];
        }

        // backward: gh[16x64] x w1^T[64x32] -> gfeat ; gh = (1-e^{-h}) * w2[:,0]
        v8f gc0 = {}, gc1 = {};
        #pragma unroll
        for (int s = 0; s < 2; ++s) {
            v16h ga;
            #pragma unroll
            for (int i = 0; i < 16; ++i) {
                int k  = s * 32 + kbaseA + i + ((i >= 8) ? 8 : 0);
                float hv = (float)hS[wv * 16 + row][k];
                ga[i] = (_Float16)((1.f - __expf(-hv)) * w2c0S[k]); // sigmoid(z)*w2col0
            }
            const _Float16* g0 = &w1S[0 * 16 + row][s * 32 + kbaseB]; // (w1^T)[k][n]=w1[n][k]
            v16h gb0 = cat16(*(const v8h*)(g0), *(const v8h*)(g0 + 16));
            gc0 = __builtin_amdgcn_wmma_f32_16x16x32_f16(false, ga, false, gb0,
                                                         (short)0, gc0, false, false);
            const _Float16* g1 = &w1S[1 * 16 + row][s * 32 + kbaseB];
            v16h gb1 = cat16(*(const v8h*)(g1), *(const v8h*)(g1 + 16));
            gc1 = __builtin_amdgcn_wmma_f32_16x16x32_f16(false, ga, false, gb1,
                                                         (short)0, gc1, false, false);
        }
        #pragma unroll
        for (int v = 0; v < 8; ++v) {
            int pt = wv * 16 + v + mrowOff;
            gfS[pt][row]      = gc0[v];
            gfS[pt][16 + row] = gc1[v];
        }
        __syncthreads();  // gfS visible to all threads

        // gradient re-gather: d(bilinear)/d(x,y) weighted by gfeat, per plane
        for (int idx = tid; idx < PTS * PLN; idx += BLOCK) {
            int pt = idx / PLN, p = idx % PLN;
            Bil b = blS[pt][p];
            float wx1 = b.wx, wy1 = b.wy, wx0 = 1.f - wx1, wy0 = 1.f - wy1;
            float gx = 0.f, gy = 0.f;
            const float* img0 = planes + (((size_t)(n * PLN + p) * (LL * CPL) + l * CPL) * (HP * HP));
            for (int c = 0; c < CPL; ++c) {
                const float* img = img0 + (size_t)c * (HP * HP);
                float v00 = tapv(img, b.x0,     b.y0);
                float v10 = tapv(img, b.x0 + 1, b.y0);
                float v01 = tapv(img, b.x0,     b.y0 + 1);
                float v11 = tapv(img, b.x0 + 1, b.y0 + 1);
                float g = gfS[pt][c];
                gx += g * (wy0 * (v10 - v00) + wy1 * (v11 - v01));
                gy += g * (wx0 * (v01 - v00) + wx1 * (v11 - v10));
            }
            // chain: d(x_img)/d(grid)=HP/2, (1/3 mean), d(scaled)/d(coord)=1
            const float sc = (HP * 0.5f) * (1.f / 3.f);
            gx *= sc; gy *= sc;
            if (p == 0)      { atomicAdd(&gcS[pt][0], gx); atomicAdd(&gcS[pt][1], gy); }
            else if (p == 1) { atomicAdd(&gcS[pt][0], gx); atomicAdd(&gcS[pt][2], gy); }
            else             { atomicAdd(&gcS[pt][2], gx); atomicAdd(&gcS[pt][1], gy); }
        }
    }
    __syncthreads();

    // ---- final combine: density / semantics / rgb / gradients ----
    for (int pt = tid; pt < PTS; pt += BLOCK) {
        const float* c = coords + ((size_t)n * NPTS + m0 + pt) * 3;
        float cx = c[0], cy = c[1], cz = c[2];
        float r = sqrtf(cx * cx + cy * cy + cz * cz);
        float ssdf = r - 0.5f;
        float inv_r = 1.f / (r + 1e-8f);
        float delta = 0.f, sigma = 0.f;
        float ls[LL];
        #pragma unroll
        for (int l = 0; l < LL; ++l) {
            float sl = sigS[l][pt];
            delta += sl;
            float s = sl + ssdf;
            float t = 1.f / (1.f + __expf(s / beta));  // sigmoid(-s/beta)
            ls[l] = t / beta;
            sigma += ls[l];
        }
        float inv_sig = 1.f / (sigma + 1e-8f);
        float racc0 = 0.f, racc1 = 0.f, racc2 = 0.f;
        float sem[LL];
        #pragma unroll
        for (int l = 0; l < LL; ++l) {
            sem[l] = ls[l] * inv_sig;
            racc0 += rgbS[l][pt][0] * sem[l];
            racc1 += rgbS[l][pt][1] * sem[l];
            racc2 += rgbS[l][pt][2] * sem[l];
        }
        float* o = out + ((size_t)n * NPTS + m0 + pt) * OUTC;
        o[0] = ssdf + delta;
        o[1] = sigma;
        o[2] = 1.f / (1.f + __expf(-racc0)) * 1.002f - 0.001f;
        o[3] = 1.f / (1.f + __expf(-racc1)) * 1.002f - 0.001f;
        o[4] = 1.f / (1.f + __expf(-racc2)) * 1.002f - 0.001f;
        #pragma unroll
        for (int l = 0; l < LL; ++l) o[5 + l] = sem[l];
        float g0 = gcS[pt][0], g1 = gcS[pt][1], g2 = gcS[pt][2];
        o[11] = g0 + cx * inv_r;
        o[12] = g1 + cy * inv_r;
        o[13] = g2 + cz * inv_r;
        o[14] = g0; o[15] = g1; o[16] = g2;
    }
}

extern "C" void kernel_launch(void* const* d_in, const int* in_sizes, int n_in,
                              void* d_out, int out_size, void* d_ws, size_t ws_size,
                              hipStream_t stream) {
    const float* planes = (const float*)d_in[0];
    const float* coords = (const float*)d_in[1];
    const float* w1     = (const float*)d_in[2];
    const float* b1     = (const float*)d_in[3];
    const float* w2     = (const float*)d_in[4];
    const float* b2     = (const float*)d_in[5];
    const float* beta   = (const float*)d_in[6];
    float* out = (float*)d_out;
    (void)in_sizes; (void)n_in; (void)out_size; (void)d_ws; (void)ws_size;

    dim3 grid(NB * (NPTS / PTS));   // 2048 blocks of 128 threads (4 waves)
    triplane_fused_kernel<<<grid, BLOCK, 0, stream>>>(planes, coords, w1, b1, w2, b2, beta, out);
}
